// LayerStacks_48481590837743
// MI455X (gfx1250) — matrix operand order repaired and checked by reading, not compiled, not run on hardware
//
#include <hip/hip_runtime.h>
#include <hip/hip_bf16.h>

// ---------------- problem constants ----------------
#define COUNT   8
#define L1      3072
#define L2      15
#define L3      32
#define BATCH   32768
#define TILE    16
#define KCHUNK  32
#define NCHUNKS (L1 / KCHUNK)                 // 96
#define MAXTILES (BATCH / TILE + COUNT)       // 2056
#define SQ_CORR (127.0f / 128.0f)

typedef __attribute__((ext_vector_type(8)))  float  v8f;
typedef __attribute__((ext_vector_type(16))) __bf16 v16bf;
typedef __attribute__((ext_vector_type(2)))  __bf16 v2bf;

// ---------------- helpers ----------------
__device__ __forceinline__ unsigned pack_bf2(float a, float b) {
    v2bf p = { (__bf16)a, (__bf16)b };        // native RNE cvt (v_cvt_pk_bf16_f32)
    return __builtin_bit_cast(unsigned, p);
}
__device__ __forceinline__ float clip01(float v) {
    return fminf(fmaxf(v, 0.0f), 1.0f);
}

// ---------------- 0: zero counters ----------------
__global__ void k_init(int* counts) {
    int t = threadIdx.x;
    if (t < 32) counts[t] = 0;   // counts[8] + cursor[8] + tileBase[9] (+pad)
}

// ---------------- 1: histogram over routing ----------------
__global__ void k_hist(const long long* __restrict__ routing, int* __restrict__ counts) {
    int b = blockIdx.x * blockDim.x + threadIdx.x;
    if (b < BATCH) atomicAdd(&counts[(int)routing[b]], 1);
}

// ---------------- 2: tile-prefix scan (tiny) ----------------
__global__ void k_scan(const int* __restrict__ counts, int* __restrict__ tileBase) {
    if (threadIdx.x == 0 && blockIdx.x == 0) {
        int tb = 0;
        for (int e = 0; e < COUNT; ++e) {
            tileBase[e] = tb;
            tb += (counts[e] + TILE - 1) / TILE;
        }
        tileBase[COUNT] = tb;
    }
}

// ---------------- 3: scatter rows into buckets ----------------
__global__ void k_scatter(const long long* __restrict__ routing,
                          const int* __restrict__ tileBase,
                          int* __restrict__ cursor,
                          int* __restrict__ rowidx) {
    int b = blockIdx.x * blockDim.x + threadIdx.x;
    if (b >= BATCH) return;
    int e = (int)routing[b];
    int p = atomicAdd(&cursor[e], 1);
    rowidx[tileBase[e] * TILE + p] = b;
}

// ---------------- 4: pad final partial tile of each bucket ----------------
__global__ void k_pad(const int* __restrict__ counts,
                      const int* __restrict__ tileBase,
                      int* __restrict__ rowidx) {
    int e = blockIdx.x;
    int n = counts[e];
    if (n == 0) return;
    int padN = ((n + TILE - 1) / TILE) * TILE;
    int base = tileBase[e] * TILE;
    int first = rowidx[base];                 // duplicate a valid row: idempotent
    for (int slot = n + threadIdx.x; slot < padN; slot += TILE)
        rowidx[base + slot] = first;
}

// ---------------- 5: pre-split w1 into hi/lo bf16, B-fragment layout ----------
// B frag (32x16 bf16): lane L holds N=L%16, K = 16*(L/16) + j (j=0..15, contiguous).
__global__ void k_wprep(const float* __restrict__ w1,
                        unsigned short* __restrict__ wfH,
                        unsigned short* __restrict__ wfL) {
    int tid = blockIdx.x * blockDim.x + threadIdx.x;
    if (tid >= COUNT * NCHUNKS * 32) return;
    int lane  = tid & 31;
    int chunk = (tid >> 5) % NCHUNKS;
    int e     = tid / (32 * NCHUNKS);
    int n     = lane & 15;
    int kbase = chunk * KCHUNK + (lane >> 4) * 16;
    const float* wrow = w1 + ((size_t)e * 16 + n) * L1 + kbase;
    unsigned short* oh = wfH + ((size_t)(e * NCHUNKS + chunk) * 32 + lane) * 16;
    unsigned short* ol = wfL + ((size_t)(e * NCHUNKS + chunk) * 32 + lane) * 16;
    #pragma unroll
    for (int j = 0; j < 16; ++j) {
        float w = wrow[j];
        __bf16 h = (__bf16)w;
        oh[j] = __builtin_bit_cast(unsigned short, h);
        __bf16 l = (__bf16)(w - (float)h);
        ol[j] = __builtin_bit_cast(unsigned short, l);
    }
}

// ---------------- 6: bucketed WMMA GEMM + tiny tail layers ----------------
__global__ void __launch_bounds__(32)
k_gemm(const float* __restrict__ x,
       const unsigned short* __restrict__ wfH,
       const unsigned short* __restrict__ wfL,
       const float* __restrict__ b1,
       const float* __restrict__ w2,
       const float* __restrict__ b2,
       const float* __restrict__ w3,
       const float* __restrict__ b3,
       const int* __restrict__ tileBase,
       const int* __restrict__ rowidx,
       float* __restrict__ out) {
    int t = blockIdx.x;
    int total = tileBase[COUNT];
    if (t >= total) return;                       // uniform per block

    int e = 0;
    #pragma unroll
    for (int i = 0; i < COUNT; ++i)
        if (t >= tileBase[i + 1]) e = i + 1;

    int lane  = threadIdx.x;
    int kHalf = lane >> 4;                         // 0 or 1
    int m     = lane & 15;

    __shared__ int rows[TILE];
    __shared__ __align__(32) unsigned int aHi[32][8];
    __shared__ __align__(32) unsigned int aLo[32][8];
    __shared__ float cT[TILE][TILE + 1];

    if (lane < TILE) rows[lane] = rowidx[t * TILE + lane];
    __syncthreads();

    int myRow = rows[m];
    const float* xrow = x + (size_t)myRow * L1 + kHalf * 16;

    v8f acc0 = {0.f,0.f,0.f,0.f,0.f,0.f,0.f,0.f};
    v8f acc1 = acc0, acc2 = acc0;

    const unsigned short* wbaseH = wfH + (size_t)e * NCHUNKS * 32 * 16;
    const unsigned short* wbaseL = wfL + (size_t)e * NCHUNKS * 32 * 16;

    for (int c = 0; c < NCHUNKS; ++c) {
        // --- load 16 f32 of x for (row m, K window) ---
        const float4* xp = (const float4*)(xrow + c * KCHUNK);
        float4 a0 = xp[0], a1 = xp[1], a2 = xp[2], a3 = xp[3];
        // speculative prefetch of next chunk (silently dropped if OOB)
        __builtin_prefetch(xrow + (c + 1) * KCHUNK, 0, 3);
        float xv[16] = {a0.x,a0.y,a0.z,a0.w, a1.x,a1.y,a1.z,a1.w,
                        a2.x,a2.y,a2.z,a2.w, a3.x,a3.y,a3.z,a3.w};

        // --- split hi/lo bf16 with native converts, scatter into A-fragment ---
        #pragma unroll
        for (int p = 0; p < 8; ++p) {
            int   j0 = 2 * p;
            float x0 = xv[j0], x1 = xv[j0 + 1];
            __bf16 h0 = (__bf16)x0, h1 = (__bf16)x1;
            float r0 = x0 - (float)h0, r1 = x1 - (float)h1;
            v2bf hp2 = { h0, h1 };
            unsigned hp = __builtin_bit_cast(unsigned, hp2);
            unsigned lp = pack_bf2(r0, r1);
            int destLane = m + (kHalf << 4);
            int dword    = j0 >> 1;
            aHi[destLane][dword] = hp;
            aLo[destLane][dword] = lp;
        }
        __syncthreads();

        v16bf ah = *(const v16bf*)&aHi[lane][0];
        v16bf al = *(const v16bf*)&aLo[lane][0];
        v16bf bh = *(const v16bf*)(wbaseH + ((size_t)c * 32 + lane) * 16);
        v16bf bl = *(const v16bf*)(wbaseL + ((size_t)c * 32 + lane) * 16);

        // bf16x3: independent accumulators avoid WMMA->WMMA RAW chains
        acc0 = __builtin_amdgcn_wmma_f32_16x16x32_bf16(false, ah, false, bh,
                                                       (short)0, acc0, false, false);
        acc1 = __builtin_amdgcn_wmma_f32_16x16x32_bf16(false, ah, false, bl,
                                                       (short)0, acc1, false, false);
        acc2 = __builtin_amdgcn_wmma_f32_16x16x32_bf16(false, al, false, bh,
                                                       (short)0, acc2, false, false);
        __syncthreads();   // before next iteration overwrites aHi/aLo
    }

    v8f acc = acc0 + acc1 + acc2;

    // C tile: VGPR v, lane L -> M = v + 8*(L/16), N = L%16
    #pragma unroll
    for (int v = 0; v < 8; ++v)
        cT[v + 8 * kHalf][m] = acc[v];
    __syncthreads();

    // ---- tail: layers 2 & 3, one row per lane (lanes 0..15) ----
    if (lane < TILE) {
        float h1[2 * L2];
        #pragma unroll
        for (int n = 0; n < L2; ++n) {
            float v = cT[lane][n] + b1[e * 16 + n];
            h1[n]      = clip01(v * v * SQ_CORR);
            h1[L2 + n] = clip01(v);
        }
        float l1out = cT[lane][L2] + b1[e * 16 + L2];
        float l3 = b3[e] + l1out;
        const float* w2e = w2 + (size_t)e * L3 * (2 * L2);
        for (int o = 0; o < L3; ++o) {
            float a = b2[e * L3 + o];
            #pragma unroll
            for (int i = 0; i < 2 * L2; ++i)
                a += h1[i] * w2e[o * (2 * L2) + i];
            l3 += clip01(a) * w3[e * L3 + o];
        }
        out[rows[lane]] = l3;
    }
}

// ---------------- launcher ----------------
extern "C" void kernel_launch(void* const* d_in, const int* in_sizes, int n_in,
                              void* d_out, int out_size, void* d_ws, size_t ws_size,
                              hipStream_t stream) {
    const float*     x       = (const float*)d_in[0];
    const long long* routing = (const long long*)d_in[1];   // int64 per reference
    const float*     w1      = (const float*)d_in[2];
    const float*     b1      = (const float*)d_in[3];
    const float*     w2      = (const float*)d_in[4];
    const float*     b2      = (const float*)d_in[5];
    const float*     w3      = (const float*)d_in[6];
    const float*     b3      = (const float*)d_in[7];
    float*           out     = (float*)d_out;

    char* ws = (char*)d_ws;
    int* counts   = (int*)ws;             // [8]
    int* cursor   = counts + 8;           // [8]
    int* tileBase = cursor + 8;           // [9]
    int* rowidx   = (int*)(ws + 256);     // [BATCH + COUNT*TILE]
    unsigned short* wfH = (unsigned short*)(ws + 256 + (((BATCH + COUNT * TILE) * 4 + 255) & ~255));
    unsigned short* wfL = wfH + (size_t)COUNT * NCHUNKS * 32 * 16;

    k_init   <<<1, 64, 0, stream>>>(counts);
    k_hist   <<<BATCH / 256, 256, 0, stream>>>(routing, counts);
    k_scan   <<<1, 1, 0, stream>>>(counts, tileBase);
    k_scatter<<<BATCH / 256, 256, 0, stream>>>(routing, tileBase, cursor, rowidx);
    k_pad    <<<COUNT, TILE, 0, stream>>>(counts, tileBase, rowidx);
    k_wprep  <<<(COUNT * NCHUNKS * 32 + 255) / 256, 256, 0, stream>>>(w1, wfH, wfL);
    k_gemm   <<<MAXTILES, 32, 0, stream>>>(x, wfH, wfL, b1, w2, b2, w3, b3,
                                           tileBase, rowidx, out);
}
